// MOME_layer_21543555957376
// MI455X (gfx1250) — matrix operand order repaired and checked by reading, not compiled, not run on hardware
//
#include <hip/hip_runtime.h>
#include <math.h>

#define NTOK 2048
#define DMODEL 1024
#define NH 4
#define HALFD 256
#define NKEY 256
#define KNN 16
#define NEXP 16
#define DOUT_ 1024
#define HOUT 2048   // H*2*HALF

#define NEG_INF (-3.0e38f)

typedef __bf16 bf16;
typedef __attribute__((ext_vector_type(16))) __bf16 bf16x16;
typedef __attribute__((ext_vector_type(8)))  float  f32x8;

union FragAB { bf16x16 v; uint4 q[2]; };
union FragC  { f32x8 v; float f[8]; };

// ---------------------------------------------------------------------------
// 1) fp32 -> bf16 elementwise convert (keys table, data)
__global__ void k_cvt_bf16(const float* __restrict__ src, bf16* __restrict__ dst, int n) {
    int i = blockIdx.x * 256 + threadIdx.x;
    if (i < n) dst[i] = (bf16)src[i];
}

// ---------------------------------------------------------------------------
// 2) fp32 [K][NO] -> bf16 transposed [NO][K]; LDS-tiled 64x64 so both global
//    sides are b128-coalesced. blockIdx.z selects matrix (stride elements).
__global__ void k_tcvt(const float* __restrict__ W, bf16* __restrict__ Wt,
                       int K, int NO, size_t mat_stride) {
    __shared__ __align__(16) bf16 Ls[64][80];   // padded rows: 160B, 16B-aligned
    const float* Wsrc = W + mat_stride * blockIdx.z;
    bf16* Wdst = Wt + mat_stride * blockIdx.z;
    int k0 = blockIdx.y * 64, n0 = blockIdx.x * 64;
    int t = threadIdx.x;  // 256
#pragma unroll
    for (int it = 0; it < 4; ++it) {
        int lin = (it * 256 + t) * 4;            // 0..4095
        int k = lin >> 6, n = lin & 63;
        float4 f = *(const float4*)(Wsrc + (size_t)(k0 + k) * NO + n0 + n);
        Ls[n + 0][k] = (bf16)f.x; Ls[n + 1][k] = (bf16)f.y;
        Ls[n + 2][k] = (bf16)f.z; Ls[n + 3][k] = (bf16)f.w;
    }
    __syncthreads();
#pragma unroll
    for (int it = 0; it < 2; ++it) {
        int lin = (it * 256 + t) * 8;            // 0..4095
        int n = lin >> 6, k = lin & 63;
        *(uint4*)(Wdst + (size_t)(n0 + n) * K + k0 + k) = *(const uint4*)&Ls[n][k];
    }
}

// ---------------------------------------------------------------------------
// 3) q = data @ W_q -> bf16 [NTOK, HOUT]; LDS-free WMMA, wave tile 16x32
__global__ void k_qgemm(const bf16* __restrict__ Xb, const bf16* __restrict__ Wt,
                        bf16* __restrict__ Q) {
    int t = threadIdx.x, lane = t & 31, wv = t >> 5;
    int mBase = blockIdx.y * 16;
    int nBase = blockIdx.x * 128 + wv * 32;
    FragC acc0, acc1;
#pragma unroll
    for (int i = 0; i < 8; ++i) { acc0.f[i] = 0.f; acc1.f[i] = 0.f; }
    int m = lane & 15, kb = (lane >> 4) * 8, kb2 = (lane >> 4) * 16;
    const bf16* arow = Xb + (size_t)(mBase + m) * DMODEL;
    const bf16* b0 = Wt + (size_t)(nBase + (lane & 15)) * DMODEL;
    const bf16* b1 = Wt + (size_t)(nBase + 16 + (lane & 15)) * DMODEL;
    for (int k0 = 0; k0 < DMODEL; k0 += 32) {
        FragAB a, bA, bB;
        a.q[0]  = *(const uint4*)(arow + k0 + kb2);
        a.q[1]  = *(const uint4*)(arow + k0 + kb2 + 8);
        bA.q[0] = *(const uint4*)(b0 + k0 + kb2);
        bA.q[1] = *(const uint4*)(b0 + k0 + kb2 + 8);
        bB.q[0] = *(const uint4*)(b1 + k0 + kb2);
        bB.q[1] = *(const uint4*)(b1 + k0 + kb2 + 8);
        acc0.v = __builtin_amdgcn_wmma_f32_16x16x32_bf16(false, a.v, false, bA.v,
                                                         (short)0, acc0.v, false, false);
        acc1.v = __builtin_amdgcn_wmma_f32_16x16x32_bf16(false, a.v, false, bB.v,
                                                         (short)0, acc1.v, false, false);
    }
    int col = nBase + (lane & 15);
    int rbase = mBase + (lane >> 4) * 8;
#pragma unroll
    for (int r = 0; r < 8; ++r) {
        Q[(size_t)(rbase + r) * HOUT + col]      = (bf16)acc0.f[r];
        Q[(size_t)(rbase + r) * HOUT + col + 16] = (bf16)acc1.f[r];
    }
}

// ---------------------------------------------------------------------------
// 4) scores[n,hc,k] = q[n,hc,:] . keys[hc,k,:]; LDS-free, wave tile 16x32
__global__ void k_scores(const bf16* __restrict__ Q, const bf16* __restrict__ Kb,
                         float* __restrict__ S) {
    int t = threadIdx.x, lane = t & 31, wv = t >> 5;
    int hc = blockIdx.z;
    int mBase = blockIdx.y * 16;
    int nBase = blockIdx.x * 128 + wv * 32;
    FragC acc0, acc1;
#pragma unroll
    for (int i = 0; i < 8; ++i) { acc0.f[i] = 0.f; acc1.f[i] = 0.f; }
    int m = lane & 15, kb = (lane >> 4) * 8, kb2 = (lane >> 4) * 16;
    const bf16* qrow = Q + (size_t)(mBase + m) * HOUT + hc * HALFD;
    int n = nBase + (lane & 15);
    const bf16* k0p = Kb + ((size_t)hc * NKEY + n) * HALFD;
    const bf16* k1p = Kb + ((size_t)hc * NKEY + n + 16) * HALFD;
    for (int k0 = 0; k0 < HALFD; k0 += 32) {
        FragAB a, bA, bB;
        a.q[0]  = *(const uint4*)(qrow + k0 + kb2);
        a.q[1]  = *(const uint4*)(qrow + k0 + kb2 + 8);
        bA.q[0] = *(const uint4*)(k0p + k0 + kb2);
        bA.q[1] = *(const uint4*)(k0p + k0 + kb2 + 8);
        bB.q[0] = *(const uint4*)(k1p + k0 + kb2);
        bB.q[1] = *(const uint4*)(k1p + k0 + kb2 + 8);
        acc0.v = __builtin_amdgcn_wmma_f32_16x16x32_bf16(false, a.v, false, bA.v,
                                                         (short)0, acc0.v, false, false);
        acc1.v = __builtin_amdgcn_wmma_f32_16x16x32_bf16(false, a.v, false, bB.v,
                                                         (short)0, acc1.v, false, false);
    }
    int rbase = mBase + (lane >> 4) * 8;
#pragma unroll
    for (int r = 0; r < 8; ++r) {
        S[((size_t)(rbase + r) * 8 + hc) * NKEY + n]      = acc0.f[r];
        S[((size_t)(rbase + r) * 8 + hc) * NKEY + n + 16] = acc1.f[r];
    }
}

// ---------------------------------------------------------------------------
// 5) top-16 of each 256-wide score row; wave per row, iterative max
__global__ void k_top1(const float* __restrict__ S, float* __restrict__ TV,
                       int* __restrict__ TI) {
    int lane = threadIdx.x & 31;
    int row = blockIdx.x * 4 + (threadIdx.x >> 5);   // N*H*2 = 16384 rows
    const float* base = S + (size_t)row * 256;
    float v[8];
#pragma unroll
    for (int j = 0; j < 8; ++j) v[j] = base[j * 32 + lane];
    for (int k = 0; k < KNN; ++k) {
        float bv = NEG_INF; int bi = 0x7fffffff;
#pragma unroll
        for (int j = 0; j < 8; ++j) {
            int idx = j * 32 + lane;
            if (v[j] > bv || (v[j] == bv && idx < bi)) { bv = v[j]; bi = idx; }
        }
        for (int off = 16; off; off >>= 1) {
            float ov = __shfl_xor(bv, off, 32);
            int   oi = __shfl_xor(bi, off, 32);
            if (ov > bv || (ov == bv && oi < bi)) { bv = ov; bi = oi; }
        }
#pragma unroll
        for (int j = 0; j < 8; ++j) if (j * 32 + lane == bi) v[j] = NEG_INF;
        if (lane == k) { TV[row * 16 + k] = bv; TI[row * 16 + k] = bi; }
    }
}

// ---------------------------------------------------------------------------
// 6) per (n,h): top-16 of the 16x16 candidate sums, softmax, value row ids
__global__ void k_top2(const float* __restrict__ TV, const int* __restrict__ TI,
                       float* __restrict__ Wsel, int* __restrict__ Vidx) {
    int lane = threadIdx.x & 31;
    int nh = blockIdx.x * 4 + (threadIdx.x >> 5);    // N*H = 8192 rows
    int n = nh >> 2, h = nh & 3;
    int row1 = n * 8 + h * 2, row2 = row1 + 1;
    float v1r = (lane < 16) ? TV[row1 * 16 + lane] : 0.f;
    float v2r = (lane < 16) ? TV[row2 * 16 + lane] : 0.f;
    float cv[8];
#pragma unroll
    for (int j = 0; j < 8; ++j) {
        int c = lane * 8 + j;
        cv[j] = __shfl(v1r, c >> 4, 32) + __shfl(v2r, c & 15, 32);
    }
    float myv = NEG_INF; int myc = 0;
    for (int k = 0; k < KNN; ++k) {
        float bv = NEG_INF; int bc = 0x7fffffff;
#pragma unroll
        for (int j = 0; j < 8; ++j) {
            int c = lane * 8 + j;
            if (cv[j] > bv || (cv[j] == bv && c < bc)) { bv = cv[j]; bc = c; }
        }
        for (int off = 16; off; off >>= 1) {
            float ov = __shfl_xor(bv, off, 32);
            int   oc = __shfl_xor(bc, off, 32);
            if (ov > bv || (ov == bv && oc < bc)) { bv = ov; bc = oc; }
        }
#pragma unroll
        for (int j = 0; j < 8; ++j) if (lane * 8 + j == bc) cv[j] = NEG_INF;
        if (lane == k) { myv = bv; myc = bc; }
    }
    float vmax = __shfl(myv, 0, 32);
    float e = (lane < 16) ? __expf(myv - vmax) : 0.f;
    float s = e;
    for (int off = 16; off; off >>= 1) s += __shfl_xor(s, off, 32);
    if (lane < 16) {
        Wsel[nh * 16 + lane] = e / s;
        int i1 = TI[row1 * 16 + (myc >> 4)];
        int i2 = TI[row2 * 16 + (myc & 15)];
        Vidx[nh * 16 + lane] = i1 * NKEY + i2;
    }
}

// ---------------------------------------------------------------------------
// 7) x = data + sum_{h,k} w * values[idx]   (~537MB gather), bf16 copy too
__global__ void k_gather(const float* __restrict__ Xin, const float* __restrict__ Vals,
                         const float* __restrict__ Wsel, const int* __restrict__ Vidx,
                         float* __restrict__ Xout, bf16* __restrict__ Xbf) {
    __shared__ float ws[64]; __shared__ int ids[64];
    int n = blockIdx.x, t = threadIdx.x;
    if (t < 64) { ws[t] = Wsel[n * 64 + t]; ids[t] = Vidx[n * 64 + t]; }
    __syncthreads();
    int c = t * 4;
    float4 a = *(const float4*)(Xin + (size_t)n * DMODEL + c);
    for (int j = 0; j < 64; ++j) {
        if (j + 1 < 64)  // prefetch next gathered row (global_prefetch_b8)
            __builtin_prefetch(Vals + (size_t)ids[j + 1] * DMODEL + c, 0, 0);
        float w = ws[j];
        float4 vv = *(const float4*)(Vals + (size_t)ids[j] * DMODEL + c);
        a.x += w * vv.x; a.y += w * vv.y; a.z += w * vv.z; a.w += w * vv.w;
    }
    *(float4*)(Xout + (size_t)n * DMODEL + c) = a;
    bf16* p = Xbf + (size_t)n * DMODEL + c;
    p[0] = (bf16)a.x; p[1] = (bf16)a.y; p[2] = (bf16)a.z; p[3] = (bf16)a.w;
}

// ---------------------------------------------------------------------------
// 8) gate logits + top-2 + softmax
__global__ void k_gate(const float* __restrict__ X, const float* __restrict__ GW,
                       const float* __restrict__ GB, int* __restrict__ gi0,
                       int* __restrict__ gi1, float* __restrict__ g0,
                       float* __restrict__ g1) {
    __shared__ __align__(16) float xs[DMODEL];
    __shared__ float red[16][8];
    __shared__ float lg[16];
    int n = blockIdx.x, t = threadIdx.x;  // 128 threads
    {
        const float4* src = (const float4*)(X + (size_t)n * DMODEL);
        ((float4*)xs)[t] = src[t];
        ((float4*)xs)[t + 128] = src[t + 128];
    }
    __syncthreads();
    int e = t >> 3, part = t & 7;
    float s = 0.f;
    for (int d = part * 128; d < part * 128 + 128; ++d) s += xs[d] * GW[d * NEXP + e];
    red[e][part] = s;
    __syncthreads();
    if (t < 16) {
        float tot = 0.f;
#pragma unroll
        for (int p = 0; p < 8; ++p) tot += red[t][p];
        lg[t] = tot + GB[t];
    }
    __syncthreads();
    if (t == 0) {
        float m0 = NEG_INF, m1 = NEG_INF; int i0 = 0, i1_ = 0;
        for (int e2 = 0; e2 < NEXP; ++e2) {
            float v = lg[e2];
            if (v > m0)      { m1 = m0; i1_ = i0; m0 = v; i0 = e2; }
            else if (v > m1) { m1 = v; i1_ = e2; }
        }
        float e1v = __expf(m1 - m0);
        gi0[n] = i0; gi1[n] = i1_;
        g0[n] = 1.f / (1.f + e1v);
        g1[n] = e1v / (1.f + e1v);
    }
}

// ---------------------------------------------------------------------------
// 9) deterministic per-expert token compaction (prefix-scan, no atomics)
__global__ void k_route(const int* __restrict__ gi0, const int* __restrict__ gi1,
                        const float* __restrict__ g0, const float* __restrict__ g1,
                        int* __restrict__ ltok, float* __restrict__ lgt,
                        int* __restrict__ counts) {
    __shared__ int sc[256]; __shared__ int sBase;
    int e = blockIdx.x, t = threadIdx.x;
    if (t == 0) sBase = 0;
    __syncthreads();
    for (int c0 = 0; c0 < NTOK; c0 += 256) {
        int n = c0 + t;
        int slot = -1;
        if (gi0[n] == e) slot = 0; else if (gi1[n] == e) slot = 1;
        int flag = (slot >= 0) ? 1 : 0;
        sc[t] = flag; __syncthreads();
        for (int off = 1; off < 256; off <<= 1) {   // inclusive scan
            int v = (t >= off) ? sc[t - off] : 0;
            __syncthreads();
            sc[t] += v;
            __syncthreads();
        }
        int base = sBase;
        if (flag) {
            int pos = base + sc[t] - 1;
            ltok[e * NTOK + pos] = n | (slot << 20);
            lgt[e * NTOK + pos] = slot ? g1[n] : g0[n];
        }
        __syncthreads();
        if (t == 0) sBase = base + sc[255];
        __syncthreads();
    }
    if (t == 0) counts[e] = sBase;
}

// ---------------------------------------------------------------------------
// 10) sparse expert GEMM; LDS-free: per-lane gathered A rows, bf16 Wt B rows
__global__ void k_expert(const bf16* __restrict__ Xbf, const bf16* __restrict__ EWt,
                         const float* __restrict__ EB, const int* __restrict__ ltok,
                         const float* __restrict__ lgt, const int* __restrict__ counts,
                         float* __restrict__ Y) {
    int e = blockIdx.z;
    int cnt = counts[e];
    int tile = blockIdx.y;
    if (tile * 16 >= cnt) return;
    int t = threadIdx.x, lane = t & 31, wv = t >> 5;
    int nBase = blockIdx.x * 128 + wv * 32;
    int m = lane & 15, kb = (lane >> 4) * 8, kb2 = (lane >> 4) * 16;

    // per-lane gathered A row (token for fragment row m)
    int sA = tile * 16 + m;
    int pkA = ltok[e * NTOK + ((sA < cnt) ? sA : 0)];
    int tokA = (sA < cnt) ? (pkA & 0xFFFF) : 0;
    const bf16* arow = Xbf + (size_t)tokA * DMODEL;
    const bf16* b0 = EWt + ((size_t)e * DOUT_ + nBase + (lane & 15)) * DMODEL;
    const bf16* b1 = EWt + ((size_t)e * DOUT_ + nBase + 16 + (lane & 15)) * DMODEL;

    FragC acc0, acc1;
#pragma unroll
    for (int i = 0; i < 8; ++i) { acc0.f[i] = 0.f; acc1.f[i] = 0.f; }
    for (int k0 = 0; k0 < DMODEL; k0 += 32) {
        FragAB a, bA, bB;
        a.q[0]  = *(const uint4*)(arow + k0 + kb2);
        a.q[1]  = *(const uint4*)(arow + k0 + kb2 + 8);
        bA.q[0] = *(const uint4*)(b0 + k0 + kb2);
        bA.q[1] = *(const uint4*)(b0 + k0 + kb2 + 8);
        bB.q[0] = *(const uint4*)(b1 + k0 + kb2);
        bB.q[1] = *(const uint4*)(b1 + k0 + kb2 + 8);
        acc0.v = __builtin_amdgcn_wmma_f32_16x16x32_bf16(false, a.v, false, bA.v,
                                                         (short)0, acc0.v, false, false);
        acc1.v = __builtin_amdgcn_wmma_f32_16x16x32_bf16(false, a.v, false, bB.v,
                                                         (short)0, acc1.v, false, false);
    }
    int col = nBase + (lane & 15);
    float bias0 = EB[e * DOUT_ + col];
    float bias1 = EB[e * DOUT_ + col + 16];
    int rtop = (lane >> 4) * 8;
#pragma unroll
    for (int r = 0; r < 8; ++r) {
        int s16 = tile * 16 + rtop + r;
        if (s16 < cnt) {
            int pk = ltok[e * NTOK + s16];
            float g = lgt[e * NTOK + s16];
            int token = pk & 0xFFFF, slot = (pk >> 20) & 1;
            float* yrow = Y + ((size_t)slot * NTOK + token) * DOUT_;
            yrow[col]      = g * (acc0.f[r] + bias0);
            yrow[col + 16] = g * (acc1.f[r] + bias1);
        }
    }
}

// ---------------------------------------------------------------------------
// 11) out = Y[0] + Y[1]
__global__ void k_combine(const float* __restrict__ Y, float* __restrict__ Out) {
    size_t i = ((size_t)blockIdx.x * 256 + threadIdx.x) * 4;
    float4 a = *(const float4*)(Y + i);
    float4 b = *(const float4*)(Y + (size_t)NTOK * DOUT_ + i);
    float4 o; o.x = a.x + b.x; o.y = a.y + b.y; o.z = a.z + b.z; o.w = a.w + b.w;
    *(float4*)(Out + i) = o;
}

// ---------------------------------------------------------------------------
extern "C" void kernel_launch(void* const* d_in, const int* in_sizes, int n_in,
                              void* d_out, int out_size, void* d_ws, size_t ws_size,
                              hipStream_t stream) {
    (void)in_sizes; (void)n_in; (void)out_size; (void)ws_size;
    const float* data     = (const float*)d_in[0];
    const float* W_q      = (const float*)d_in[1];
    const float* keys     = (const float*)d_in[2];
    const float* values   = (const float*)d_in[3];
    const float* gate_W   = (const float*)d_in[4];
    const float* gate_b   = (const float*)d_in[5];
    const float* expert_W = (const float*)d_in[6];
    const float* expert_b = (const float*)d_in[7];

    char* w = (char*)d_ws;
    size_t off = 0;
    auto alloc = [&](size_t bytes) -> void* {
        void* p = w + off;
        off += (bytes + 255) & ~(size_t)255;
        return p;
    };
    bf16*  data_bf = (bf16*) alloc((size_t)NTOK * DMODEL * 2);          //  4 MB
    bf16*  Wqt     = (bf16*) alloc((size_t)HOUT * DMODEL * 2);          //  4 MB
    bf16*  EWt     = (bf16*) alloc((size_t)NEXP * DOUT_ * DMODEL * 2);  // 32 MB
    bf16*  q_bf    = (bf16*) alloc((size_t)NTOK * HOUT * 2);            //  8 MB
    float* scores  = (float*)alloc((size_t)NTOK * 8 * NKEY * 4);        // 16 MB
    bf16*  keys_bf = (bf16*) alloc((size_t)NH * 2 * NKEY * HALFD * 2);  //  1 MB
    float* TV      = (float*)alloc((size_t)NTOK * 8 * 16 * 4);
    int*   TI      = (int*)  alloc((size_t)NTOK * 8 * 16 * 4);
    float* Wsel    = (float*)alloc((size_t)NTOK * NH * 16 * 4);
    int*   Vidx    = (int*)  alloc((size_t)NTOK * NH * 16 * 4);
    float* Xf      = (float*)alloc((size_t)NTOK * DMODEL * 4);          //  8 MB
    bf16*  Xbf     = (bf16*) alloc((size_t)NTOK * DMODEL * 2);          //  4 MB
    int*   gi0     = (int*)  alloc((size_t)NTOK * 4);
    int*   gi1     = (int*)  alloc((size_t)NTOK * 4);
    float* g0      = (float*)alloc((size_t)NTOK * 4);
    float* g1      = (float*)alloc((size_t)NTOK * 4);
    int*   ltok    = (int*)  alloc((size_t)NEXP * NTOK * 4);
    float* lgt     = (float*)alloc((size_t)NEXP * NTOK * 4);
    int*   counts  = (int*)  alloc(256);
    float* Y       = (float*)alloc((size_t)2 * NTOK * DOUT_ * 4);       // 16 MB

    // prep: conversions & weight transposes (one-shot, off the critical loops)
    k_cvt_bf16<<<(NH * 2 * NKEY * HALFD + 255) / 256, 256, 0, stream>>>(
        keys, keys_bf, NH * 2 * NKEY * HALFD);
    k_cvt_bf16<<<(NTOK * DMODEL + 255) / 256, 256, 0, stream>>>(
        data, data_bf, NTOK * DMODEL);
    k_tcvt<<<dim3(HOUT / 64, DMODEL / 64, 1), 256, 0, stream>>>(
        W_q, Wqt, DMODEL, HOUT, 0);
    k_tcvt<<<dim3(DOUT_ / 64, DMODEL / 64, NEXP), 256, 0, stream>>>(
        expert_W, EWt, DMODEL, DOUT_, (size_t)DMODEL * DOUT_);

    // PKM
    k_qgemm<<<dim3(HOUT / 128, NTOK / 16), 128, 0, stream>>>(data_bf, Wqt, q_bf);
    k_scores<<<dim3(NKEY / 128, NTOK / 16, 8), 128, 0, stream>>>(q_bf, keys_bf, scores);
    k_top1<<<(NTOK * 8) / 4, 128, 0, stream>>>(scores, TV, TI);
    k_top2<<<(NTOK * NH) / 4, 128, 0, stream>>>(TV, TI, Wsel, Vidx);
    k_gather<<<NTOK, 256, 0, stream>>>(data, values, Wsel, Vidx, Xf, Xbf);

    // MoE
    k_gate<<<NTOK, 128, 0, stream>>>(Xf, gate_W, gate_b, gi0, gi1, g0, g1);
    k_route<<<NEXP, 256, 0, stream>>>(gi0, gi1, g0, g1, ltok, lgt, counts);
    k_expert<<<dim3(DOUT_ / 128, NTOK / 16, NEXP), 128, 0, stream>>>(
        Xbf, EWt, expert_b, ltok, lgt, counts, Y);
    k_combine<<<(NTOK * DOUT_) / 4 / 256, 256, 0, stream>>>(Y, (float*)d_out);
}
